// OutputNetwork_78718160601110
// MI455X (gfx1250) — matrix operand order repaired and checked by reading, not compiled, not run on hardware
//
#include <hip/hip_runtime.h>
#include <math.h>

typedef __attribute__((ext_vector_type(16))) _Float16 v16h;
typedef __attribute__((ext_vector_type(8)))  float    v8f;

#define BATCH   64
#define ATOMS   64
#define DIMIN   240
#define HDIM    128
#define ODIM    112

// d_ws layout (in _Float16 units unless noted)
#define W1_OFF 0        // 8 frags  * 512 halves
#define W2_OFF 4096     // 32 frags * 512 halves
#define W3_OFF 20480    // 28 frags * 512 halves
#define INV_BYTE_OFF 69632   // then 64 floats: 1/sqrt(n_atoms) per batch

#define HSTRIDE 136     // padded hidden row stride (halves) -> conflict-free b128 LDS reads
#define RSTRIDE 113     // padded R row stride (floats)

// Branch-free shifted softplus using HW transcendentals (v_exp_f32/v_log_f32).
// softplus(t) = max(t,0) + log(1 + exp(-|t|)); ssp = (softplus(5x) - ln2)/5
__device__ __forceinline__ float ssp_f(float x) {
  float t  = 5.0f * x;
  float sp = fmaxf(t, 0.0f) + __logf(1.0f + __expf(-fabsf(t)));
  return (sp - 0.69314718055994531f) * 0.2f;
}

// Pack weights into wave32 WMMA B-fragment layout, fold all static scales.
__global__ __launch_bounds__(128) void prep_kernel(
    const float* __restrict__ W1, const float* __restrict__ W2,
    const float* __restrict__ W3, const float* __restrict__ mask,
    _Float16* __restrict__ wsH, float* __restrict__ invsq)
{
  int u = blockIdx.x * blockDim.x + threadIdx.x;
  const float s1 = 0.17677669529663687f;   // 1/sqrt(32)
  const float s2 = 0.08838834764831843f;   // 1/sqrt(128)
  if (u < 256) {                            // W1: 1 K-tile, 8 N-tiles
    int f = u >> 5, l = u & 31;
    for (int e = 0; e < 16; ++e) {
      int k = ((l >> 4) << 4) + e;          // K_total = 32
      int n = f * 16 + (l & 15);
      wsH[W1_OFF + (f * 32 + l) * 16 + e] = (_Float16)(W1[k * HDIM + n] * s1);
    }
  } else if (u < 1280) {                    // W2: 4 K-tiles x 8 N-tiles
    int uu = u - 256; int f = uu >> 5, l = uu & 31;
    int kt = f >> 3, nt = f & 7;
    for (int e = 0; e < 16; ++e) {
      int k = kt * 32 + ((l >> 4) << 4) + e;
      int n = nt * 16 + (l & 15);
      wsH[W2_OFF + (f * 32 + l) * 16 + e] = (_Float16)(W2[k * HDIM + n] * s2);
    }
  } else if (u < 2176) {                    // W3: 4 K-tiles x 7 N-tiles, fold l-norms
    int uu = u - 1280; int f = uu >> 5, l = uu & 31;
    int kt = f / 7, nt = f % 7;
    for (int e = 0; e < 16; ++e) {
      int k = kt * 32 + ((l >> 4) << 4) + e;
      int n = nt * 16 + (l & 15);
      float norm = (n < 64) ? 0.125f                      // 1/sqrt(64)
                 : (n < 96) ? 0.10206207261596575f        // 1/sqrt(96)
                            : 0.11180339887498948f;       // 1/sqrt(80)
      wsH[W3_OFF + (f * 32 + l) * 16 + e] = (_Float16)(W3[k * ODIM + n] * s2 * norm);
    }
  } else if (u < 2240) {                    // 1/sqrt(n_atoms) per batch
    int z = u - 2176;
    float s = 0.0f;
    for (int j = 0; j < ATOMS; ++j) s += mask[z * ATOMS + j];
    invsq[z] = rsqrtf(fmaxf(s, 1.0f));
  }
}

// One workgroup per (z, a); 4 waves, each wave owns 16 neighbors b.
__global__ __launch_bounds__(128) void conv_kernel(
    const float* __restrict__ rep, const float* __restrict__ geom,
    const float* __restrict__ mask,
    const _Float16* __restrict__ wsH, const float* __restrict__ invsq,
    float* __restrict__ out)
{
  __shared__ __align__(16) _Float16 s_hid[4][16 * HSTRIDE];
  __shared__ float s_R[4][16 * RSTRIDE];
  __shared__ float s_acc;

  const int a = blockIdx.x, z = blockIdx.y;
  const int tid  = threadIdx.x;
  const int wave = tid >> 5, lane = tid & 31;
  const int m = lane & 15;   // tile row (pair index within wave's 16 neighbors)
  const int h = lane >> 4;   // lane half-group
  const int b = wave * 16 + m;

  if (tid == 0) s_acc = 0.0f;
  __syncthreads();

  // ---- pair geometry + real spherical harmonics -------------------------
  const float* ga = geom + (z * ATOMS + a) * 3;
  const float* gb = geom + (z * ATOMS + b) * 3;
  float rx = ga[0] - gb[0], ry = ga[1] - gb[1], rz = ga[2] - gb[2];
  float r2 = rx * rx + ry * ry + rz * rz;
  float r  = sqrtf(fmaxf(r2, 1e-12f));
  float nzf = (r2 > 1e-10f) ? 1.0f : 0.0f;
  float ir = 1.0f / r;
  float x = rx * ir, y = ry * ir, zc = rz * ir;
  const float s3 = 1.7320508075688772f, s15 = 3.872983346207417f, s5 = 2.23606797749979f;
  float y1x = s3 * x * nzf, y1y = s3 * y * nzf, y1z = s3 * zc * nzf;
  float y2a = s15 * x * y * nzf, y2b = s15 * y * zc * nzf,
        y2c = 0.5f * s5 * (3.0f * zc * zc - 1.0f) * nzf,
        y2d = s15 * x * zc * nzf,
        y2e = 0.5f * s15 * (x * x - y * y) * nzf;

  // ---- cosine basis directly in A-fragment layout (16x32 f16) -----------
  const float step = 10.0f / 31.0f, istep = 3.1f, hpi = 1.5707963267948966f;
  v16h afrag;
  #pragma unroll
  for (int e = 0; e < 16; ++e) {
    int v = e >> 1, j = e & 1;
    int k = h * 16 + v * 2 + j;
    float d = (r - (float)k * step) * istep;
    float val = (fabsf(d) < 1.0f) ? __cosf(hpi * d) : 0.0f;
    afrag[e] = (_Float16)val;
  }

  _Float16* hid = s_hid[wave];

  // ---- layer 1: (16x32) @ (32x128) --------------------------------------
  #pragma unroll
  for (int t = 0; t < 8; ++t) {
    v16h bfrag = *(const v16h*)(wsH + W1_OFF + (t * 32 + lane) * 16);
    v8f c = {};
    c = __builtin_amdgcn_wmma_f32_16x16x32_f16(false, afrag, false, bfrag,
                                               (short)0, c, false, false);
    #pragma unroll
    for (int p = 0; p < 8; ++p)
      hid[(p + h * 8) * HSTRIDE + t * 16 + (lane & 15)] = (_Float16)ssp_f(c[p]);
  }

  // re-gather hidden1 into A-fragment registers (2x ds_load_b128 per K-tile)
  v16h hfrag[4];
  #pragma unroll
  for (int kt = 0; kt < 4; ++kt) {
    #pragma unroll
    for (int e = 0; e < 8; ++e) {
      hfrag[kt][e]     = hid[m * HSTRIDE + kt * 32 + h * 16 + e];
      hfrag[kt][8 + e] = hid[m * HSTRIDE + kt * 32 + h * 16 + 8 + e];
    }
  }

  // ---- layer 2: (16x128) @ (128x128) ------------------------------------
  #pragma unroll
  for (int t = 0; t < 8; ++t) {
    v8f c = {};
    #pragma unroll
    for (int kt = 0; kt < 4; ++kt) {
      v16h bfrag = *(const v16h*)(wsH + W2_OFF + ((kt * 8 + t) * 32 + lane) * 16);
      c = __builtin_amdgcn_wmma_f32_16x16x32_f16(false, hfrag[kt], false, bfrag,
                                                 (short)0, c, false, false);
    }
    #pragma unroll
    for (int p = 0; p < 8; ++p)
      hid[(p + h * 8) * HSTRIDE + t * 16 + (lane & 15)] = (_Float16)ssp_f(c[p]);
  }

  #pragma unroll
  for (int kt = 0; kt < 4; ++kt) {
    #pragma unroll
    for (int e = 0; e < 8; ++e) {
      hfrag[kt][e]     = hid[m * HSTRIDE + kt * 32 + h * 16 + e];
      hfrag[kt][8 + e] = hid[m * HSTRIDE + kt * 32 + h * 16 + 8 + e];
    }
  }

  // ---- layer 3: (16x128) @ (128x112), norms pre-folded ------------------
  float* Rb = s_R[wave];
  #pragma unroll
  for (int t = 0; t < 7; ++t) {
    v8f c = {};
    #pragma unroll
    for (int kt = 0; kt < 4; ++kt) {
      v16h bfrag = *(const v16h*)(wsH + W3_OFF + ((kt * 7 + t) * 32 + lane) * 16);
      c = __builtin_amdgcn_wmma_f32_16x16x32_f16(false, hfrag[kt], false, bfrag,
                                                 (short)0, c, false, false);
    }
    #pragma unroll
    for (int p = 0; p < 8; ++p)
      Rb[(p + h * 8) * RSTRIDE + t * 16 + (lane & 15)] = c[p];
  }

  // ---- SH x feature contraction (straight-line, 3 l-segments) -----------
  // lane handles row m, channels c = h, h+2, ... (stride 2 for balance)
  const float* frow = rep + (z * ATOMS + b) * DIMIN;
  const float* Rrow = Rb + m * RSTRIDE;
  float part = 0.0f;
  #pragma unroll
  for (int ci = 0; ci < 32; ++ci) {          // l=0: c in [0,64)
    int c = h + 2 * ci;
    part += Rrow[c] * frow[c];
  }
  #pragma unroll
  for (int ci = 0; ci < 16; ++ci) {          // l=1: c in [64,96)
    int c = 64 + h + 2 * ci;
    const float* f = frow + 64 + 3 * (c - 64);
    part += Rrow[c] * (y1x * f[0] + y1y * f[1] + y1z * f[2]);
  }
  #pragma unroll
  for (int ci = 0; ci < 8; ++ci) {           // l=2: c in [96,112)
    int c = 96 + h + 2 * ci;
    const float* f = frow + 160 + 5 * (c - 96);
    part += Rrow[c] * (y2a * f[0] + y2b * f[1] + y2c * f[2] + y2d * f[3] + y2e * f[4]);
  }
  part *= invsq[z];  // feats = representation / sqrt(n_atoms)

  // wave32 butterfly reduction, then block accumulate
  #pragma unroll
  for (int off = 16; off > 0; off >>= 1)
    part += __shfl_xor(part, off, 32);
  if (lane == 0) atomicAdd(&s_acc, part);
  __syncthreads();

  if (tid == 0) {
    float t5 = 5.0f * s_acc;
    float sp = fmaxf(t5, 0.0f) + __logf(1.0f + __expf(-fabsf(t5)));
    out[z * ATOMS + a] = (sp * 0.2f) * mask[z * ATOMS + a];
  }
}

extern "C" void kernel_launch(void* const* d_in, const int* in_sizes, int n_in,
                              void* d_out, int out_size, void* d_ws, size_t ws_size,
                              hipStream_t stream)
{
  (void)in_sizes; (void)n_in; (void)out_size; (void)ws_size;
  const float* rep  = (const float*)d_in[0];
  const float* geom = (const float*)d_in[1];
  const float* mask = (const float*)d_in[2];
  const float* W1   = (const float*)d_in[3];
  const float* W2   = (const float*)d_in[4];
  const float* W3   = (const float*)d_in[5];
  _Float16* wsH  = (_Float16*)d_ws;
  float*    invs = (float*)((char*)d_ws + INV_BYTE_OFF);
  float*    out  = (float*)d_out;

  prep_kernel<<<dim3(18), dim3(128), 0, stream>>>(W1, W2, W3, mask, wsH, invs);
  conv_kernel<<<dim3(ATOMS, BATCH), dim3(128), 0, stream>>>(rep, geom, mask, wsH, invs, out);
}
